// DifferentiableRUDY_77335181132477
// MI455X (gfx1250) — matrix operand order repaired and checked by reading, not compiled, not run on hardware
//
#include <hip/hip_runtime.h>

typedef __attribute__((ext_vector_type(16))) _Float16 v16h;
typedef __attribute__((ext_vector_type(8)))  _Float16 v8h;
typedef __attribute__((ext_vector_type(8)))  float    v8f;

#define GRID_N   64
#define B_SZ     4
#define V_SZ     512
#define N_NETS   2000
#define MAXP     8
#define KPAD     2048
#define K_SHARP  2.0f

__device__ __forceinline__ float sigf(float t) {
    return 1.0f / (1.0f + __expf(-t));
}

// ---------------------------------------------------------------------------
// Kernel 1: per (b, net) bbox + separable indicator factors.
//   Yt[b][y][net] = in_y(y)            (f16, K-contiguous rows of length KPAD)
//   Xt[b][x][net] = in_x(x) / bbox     (f16)
// Nets in [N_NETS, KPAD) are zero-filled so WMMA K-padding adds nothing.
// ---------------------------------------------------------------------------
__global__ void rudy_setup_kernel(
    const float* __restrict__ positions,    // (B, V, 2)
    const float* __restrict__ pin_offsets,  // (P, 2)
    const int*   __restrict__ net_to_pin,   // (N, MAXP)
    const int*   __restrict__ pin_to_macro, // (P,)
    _Float16* __restrict__ Yt,              // (B, 64, KPAD)
    _Float16* __restrict__ Xt)              // (B, 64, KPAD)
{
    int tid = blockIdx.x * blockDim.x + threadIdx.x;
    if (tid >= B_SZ * KPAD) return;
    int b = tid / KPAD;
    int n = tid - b * KPAD;

    _Float16* yrow = Yt + (size_t)b * GRID_N * KPAD + n;
    _Float16* xrow = Xt + (size_t)b * GRID_N * KPAD + n;

    if (n >= N_NETS) {
        for (int g = 0; g < GRID_N; ++g) {
            yrow[g * KPAD] = (_Float16)0.0f;
            xrow[g * KPAD] = (_Float16)0.0f;
        }
        return;
    }

    // bbox over valid pins (same +/-1e9 sentinels as the reference)
    float minx = 1e9f, maxx = -1e9f, miny = 1e9f, maxy = -1e9f;
    #pragma unroll
    for (int k = 0; k < MAXP; ++k) {
        int idx = net_to_pin[n * MAXP + k];
        if (idx >= 0) {
            int m = pin_to_macro[idx];
            float px = positions[((size_t)b * V_SZ + m) * 2 + 0] + pin_offsets[idx * 2 + 0];
            float py = positions[((size_t)b * V_SZ + m) * 2 + 1] + pin_offsets[idx * 2 + 1];
            minx = fminf(minx, px); maxx = fmaxf(maxx, px);
            miny = fminf(miny, py); maxy = fmaxf(maxy, py);
        }
    }
    const float s = 0.5f * (float)(GRID_N - 1);
    float xmin = (minx + 1.0f) * s, xmax = (maxx + 1.0f) * s;
    float ymin = (miny + 1.0f) * s, ymax = (maxy + 1.0f) * s;
    float bbox = fmaxf((xmax - xmin + 1.0f) * (ymax - ymin + 1.0f), 1.0f);
    float inv_bbox = 1.0f / bbox;

    for (int g = 0; g < GRID_N; ++g) {
        float gf = (float)g;
        float in_x = sigf(K_SHARP * (gf - xmin + 0.5f)) * sigf(K_SHARP * (xmax - gf + 0.5f));
        float in_y = sigf(K_SHARP * (gf - ymin + 0.5f)) * sigf(K_SHARP * (ymax - gf + 0.5f));
        xrow[g * KPAD] = (_Float16)(in_x * inv_bbox);
        yrow[g * KPAD] = (_Float16)in_y;
    }
}

// ---------------------------------------------------------------------------
// Kernel 2: rudy[b] (64x64) = Yt[b] (64xK) x Xt[b]^T (Kx64) via WMMA f16->f32.
// 16 blocks = (b, my), 4 waves/block = nx tile column; K-loop in steps of 32.
// Fragment addressing follows CDNA5 ISA VGPR layouts:
//   A (16x32 f16): lanes 0-15 row=lane, halves K=[0..8)+[16..24);
//                  lanes 16-31 row=lane-16, halves K=[8..16)+[24..32).
//   B (32x16 f16): col=lane%16; same K halves as A: lanes 0-15 K=[0..8)+[16..24), lanes 16-31 K=[8..16)+[24..32).
//   C/D (16x16 f32): vgpr r -> row = r + (lane>=16 ? 8 : 0), col = lane%16.
// ---------------------------------------------------------------------------
__global__ void rudy_gemm_kernel(const _Float16* __restrict__ Yt,
                                 const _Float16* __restrict__ Xt,
                                 float* __restrict__ rudy)   // (B, 64, 64)
{
    int blk  = blockIdx.x;          // b*4 + my
    int b    = blk >> 2;
    int my   = blk & 3;
    int wave = threadIdx.x >> 5;    // nx tile
    int lane = threadIdx.x & 31;
    int l16  = lane & 15;
    bool hi  = (lane >= 16);

    const _Float16* arow = Yt + (size_t)(b * GRID_N + my * 16 + l16) * KPAD;
    const _Float16* brow = Xt + (size_t)(b * GRID_N + wave * 16 + l16) * KPAD;
    const int aoff = hi ? 8 : 0;    // halves
    const int boff = hi ? 8 : 0;    // halves

    v8f acc = {};
    for (int k = 0; k < KPAD; k += 32) {
        v8h a0 = *(const v8h*)(arow + k + aoff);
        v8h a1 = *(const v8h*)(arow + k + aoff + 16);
        v8h b0 = *(const v8h*)(brow + k + boff);
        v8h b1 = *(const v8h*)(brow + k + boff + 16);
        v16h av = __builtin_shufflevector(a0, a1, 0,1,2,3,4,5,6,7,8,9,10,11,12,13,14,15);
        v16h bv = __builtin_shufflevector(b0, b1, 0,1,2,3,4,5,6,7,8,9,10,11,12,13,14,15);
        acc = __builtin_amdgcn_wmma_f32_16x16x32_f16(
            /*neg_a=*/false, av, /*neg_b=*/false, bv,
            /*c_mod=*/(short)0, acc, /*reuse_a=*/false, /*reuse_b=*/false);
    }

    float* out = rudy + (size_t)b * GRID_N * GRID_N;
    int row_base = my * 16 + (hi ? 8 : 0);
    int col = wave * 16 + l16;
    #pragma unroll
    for (int r = 0; r < 8; ++r)
        out[(row_base + r) * GRID_N + col] = acc[r];
}

// ---------------------------------------------------------------------------
// Kernel 3: separable 7x7 Gaussian (sigma=1.5, zero pad) + overflow penalty.
// One block per batch, 256 threads (8 wave32), 16 pixels/thread, LDS staging.
// d_out layout: [0:4] penalty, [4:4+B*4096] rudy_smooth.
// ---------------------------------------------------------------------------
__global__ void rudy_smooth_kernel(const float* __restrict__ rudy,
                                   float* __restrict__ out)
{
    __shared__ float tile[GRID_N * GRID_N];
    __shared__ float tmpb[GRID_N * GRID_N];
    __shared__ float wred[8];

    int b = blockIdx.x;
    int tid = threadIdx.x;

    // normalized 1D Gaussian weights (2D kernel = outer product, symmetric)
    float w[7];
    {
        float sum = 0.0f;
        #pragma unroll
        for (int i = 0; i < 7; ++i) {
            float x = (float)(i - 3);
            w[i] = __expf(-x * x / (2.0f * 1.5f * 1.5f));
            sum += w[i];
        }
        float inv = 1.0f / sum;
        #pragma unroll
        for (int i = 0; i < 7; ++i) w[i] *= inv;
    }

    const float* src = rudy + (size_t)b * 4096;
    #pragma unroll
    for (int t = 0; t < 16; ++t) tile[tid + t * 256] = src[tid + t * 256];
    __syncthreads();

    // horizontal pass
    #pragma unroll 1
    for (int t = 0; t < 16; ++t) {
        int p = tid + t * 256;
        int y = p >> 6, x = p & 63;
        float s = 0.0f;
        #pragma unroll
        for (int d = -3; d <= 3; ++d) {
            int xx = x + d;
            if (xx >= 0 && xx < GRID_N) s += w[d + 3] * tile[(y << 6) + xx];
        }
        tmpb[p] = s;
    }
    __syncthreads();

    // vertical pass + overflow accumulation
    float acc = 0.0f;
    float* osm = out + 4 + (size_t)b * 4096;
    #pragma unroll 1
    for (int t = 0; t < 16; ++t) {
        int p = tid + t * 256;
        int y = p >> 6, x = p & 63;
        float s = 0.0f;
        #pragma unroll
        for (int d = -3; d <= 3; ++d) {
            int yy = y + d;
            if (yy >= 0 && yy < GRID_N) s += w[d + 3] * tmpb[(yy << 6) + x];
        }
        osm[p] = s;
        float ov = fmaxf(s - 1.0f, 0.0f);
        acc += ov * ov;
    }

    // wave32 reduction, then cross-wave via LDS (deterministic order)
    #pragma unroll
    for (int off = 16; off > 0; off >>= 1)
        acc += __shfl_down(acc, off, 32);
    if ((tid & 31) == 0) wred[tid >> 5] = acc;
    __syncthreads();
    if (tid == 0) {
        float s = 0.0f;
        #pragma unroll
        for (int i = 0; i < 8; ++i) s += wred[i];
        out[b] = s;
    }
}

extern "C" void kernel_launch(void* const* d_in, const int* in_sizes, int n_in,
                              void* d_out, int out_size, void* d_ws, size_t ws_size,
                              hipStream_t stream) {
    const float* positions    = (const float*)d_in[0];
    const float* pin_offsets  = (const float*)d_in[1];
    const int*   net_to_pin   = (const int*)d_in[2];
    const int*   pin_to_macro = (const int*)d_in[3];

    char* ws = (char*)d_ws;
    const size_t mat_bytes = (size_t)B_SZ * GRID_N * KPAD * sizeof(_Float16); // 1 MB each
    _Float16* Yt  = (_Float16*)ws;
    _Float16* Xt  = (_Float16*)(ws + mat_bytes);
    float*    rdy = (float*)(ws + 2 * mat_bytes);
    float*    out = (float*)d_out;

    rudy_setup_kernel<<<(B_SZ * KPAD + 255) / 256, 256, 0, stream>>>(
        positions, pin_offsets, net_to_pin, pin_to_macro, Yt, Xt);
    rudy_gemm_kernel<<<B_SZ * 4, 128, 0, stream>>>(Yt, Xt, rdy);
    rudy_smooth_kernel<<<B_SZ, 256, 0, stream>>>(rdy, out);
}
